// MicroMamba_38749194944669
// MI455X (gfx1250) — matrix operand order repaired and checked by reading, not compiled, not run on hardware
//
#include <hip/hip_runtime.h>
#include <hip/hip_bf16.h>
#include <math.h>

// ---------------- model constants ----------------
#define VOCAB      32000
#define DIM        768
#define DEPTH      4
#define D_STATE    16
#define D_CONV     4
#define D_INNER    1536          // DIM*EXPAND
#define DT_RANK    48            // ceil(768/16)
#define DT_RANK_P  64            // padded to WMMA K multiple
#define BATCH      2
#define SEQ        2048
#define NTOK       (BATCH*SEQ)   // 4096
#define XDIM       (2*D_INNER)   // 3072
#define DBLDIM     (DT_RANK + 2*D_STATE)  // 80
#define LN_EPS     1e-5f

typedef _Float16 h16;
typedef __attribute__((ext_vector_type(16))) _Float16 v16h;
typedef __attribute__((ext_vector_type(8)))  _Float16 v8h;
typedef __attribute__((ext_vector_type(8)))  float    v8f;
typedef int v4i __attribute__((vector_size(16)));   // 16B payload type for async DMA

#define AS1 __attribute__((address_space(1)))
#define AS3 __attribute__((address_space(3)))

// async-to-LDS DMA path (gfx1250): guarded, with sync fallback
#if defined(__has_builtin)
#  if __has_builtin(__builtin_amdgcn_global_load_async_to_lds_b128)
#    define HAVE_ASYNC_LDS 1
#  else
#    define HAVE_ASYNC_LDS 0
#  endif
#  if __has_builtin(__builtin_amdgcn_s_wait_asynccnt)
#    define WAIT_ASYNC(n) __builtin_amdgcn_s_wait_asynccnt(n)
#  else
#    define WAIT_ASYNC(n) asm volatile("s_wait_asynccnt %0" :: "i"(n) : "memory")
#  endif
#else
#  define HAVE_ASYNC_LDS 0
#  define WAIT_ASYNC(n)
#endif

__device__ __forceinline__ float sigmoid_f(float x) { return 1.0f / (1.0f + __expf(-x)); }
__device__ __forceinline__ float silu_f(float x)    { return x * sigmoid_f(x); }

// =====================================================================
// Generic WMMA GEMM:  C[M,N] = A[M,Kp] * B[N,Kp]^T   (f16 in, f32 acc)
// Block = 128 threads (4 waves), block tile 64x64, wave tile 32x32.
// Kp must be a multiple of 32.  Ragged N handled by clamping the source
// row (garbage columns are discarded by the store guard), which keeps
// the per-wave async issue count uniform (needed for s_wait_asynccnt N).
// Double-buffered GLOBAL_LOAD_ASYNC_TO_LDS_B128 pipeline when available.
// =====================================================================
__global__ void __launch_bounds__(128)
wmma_gemm_kernel(const h16* __restrict__ A, const h16* __restrict__ B,
                 float* __restrict__ C, h16* __restrict__ Ch,
                 int M, int N, int Kp)
{
    __shared__ __align__(16) h16 sA[2][64][40];   // 32-wide K stage, padded stride
    __shared__ __align__(16) h16 sB[2][64][40];

    const int tid  = threadIdx.x;
    const int lane = tid & 31;
    const int wave = tid >> 5;            // 0..3
    const int wm   = (wave >> 1) * 32;    // wave row origin in tile
    const int wn   = (wave & 1)  * 32;    // wave col origin in tile
    const int l16  = lane & 15;
    const int lh   = lane >> 4;           // 0 or 1 (lane half)

    const int tileM = blockIdx.y * 64;
    const int tileN = blockIdx.x * 64;

    // cooperative loader mapping: each thread moves 16 halves (2 x b128)
    const int ldr = tid >> 1;             // row 0..63
    const int ldc = (tid & 1) * 16;       // 0 or 16 within 32-wide K stage

    // clamped source rows (OOB columns discarded at the store)
    int am = tileM + ldr; if (am > M - 1) am = M - 1;
    int bn = tileN + ldr; if (bn > N - 1) bn = N - 1;
    const h16* Arow = A + (size_t)am * Kp + ldc;
    const h16* Brow = B + (size_t)bn * Kp + ldc;

    v8f acc[2][2];
    const v8f vzero = {0.f,0.f,0.f,0.f,0.f,0.f,0.f,0.f};
    acc[0][0] = vzero; acc[0][1] = vzero; acc[1][0] = vzero; acc[1][1] = vzero;

    const int nstage = Kp >> 5;

    auto compute = [&](int buf) {
        union FA { v16h v; v8h h[2]; };
        // A 16x32 f16: lanes 0-15 hold K {0-7,16-23}; lanes 16-31 hold K {8-15,24-31}
        FA afrag[2];
        const int kbA = lh * 8;
        #pragma unroll
        for (int i = 0; i < 2; ++i) {
            int m = wm + i * 16 + l16;
            afrag[i].h[0] = *reinterpret_cast<const v8h*>(&sA[buf][m][kbA]);
            afrag[i].h[1] = *reinterpret_cast<const v8h*>(&sA[buf][m][kbA + 16]);
        }
        // B 32x16 f16: lanes 0-15 hold K {0-7,16-23}; lanes 16-31 hold K {8-15,24-31} (N = lane%16)
        FA bfrag[2];
        const int kbB = lh * 8;
        #pragma unroll
        for (int j = 0; j < 2; ++j) {
            int n = wn + j * 16 + l16;
            bfrag[j].h[0] = *reinterpret_cast<const v8h*>(&sB[buf][n][kbB]);
            bfrag[j].h[1] = *reinterpret_cast<const v8h*>(&sB[buf][n][kbB + 16]);
        }
        #pragma unroll
        for (int i = 0; i < 2; ++i)
            #pragma unroll
            for (int j = 0; j < 2; ++j)
                acc[i][j] = __builtin_amdgcn_wmma_f32_16x16x32_f16(
                    /*neg_a=*/false, afrag[i].v, /*neg_b=*/false, bfrag[j].v,
                    /*c_mod=*/(short)0, acc[i][j], /*reuse_a=*/false, /*reuse_b=*/false);
    };

#if HAVE_ASYNC_LDS
    // 4 async DMA ops per thread per stage (uniform across the wave)
    auto issue = [&](int stage, int buf) {
        const h16* ga = Arow + (size_t)stage * 32;
        const h16* gb = Brow + (size_t)stage * 32;
        __builtin_amdgcn_global_load_async_to_lds_b128(
            (AS1 v4i*)ga,       (AS3 v4i*)&sA[buf][ldr][ldc],     0, 0);
        __builtin_amdgcn_global_load_async_to_lds_b128(
            (AS1 v4i*)(ga + 8), (AS3 v4i*)&sA[buf][ldr][ldc + 8], 0, 0);
        __builtin_amdgcn_global_load_async_to_lds_b128(
            (AS1 v4i*)gb,       (AS3 v4i*)&sB[buf][ldr][ldc],     0, 0);
        __builtin_amdgcn_global_load_async_to_lds_b128(
            (AS1 v4i*)(gb + 8), (AS3 v4i*)&sB[buf][ldr][ldc + 8], 0, 0);
    };

    issue(0, 0);
    for (int s = 0; s < nstage; ++s) {
        const int cur = s & 1;
        if (s + 1 < nstage) {
            issue(s + 1, cur ^ 1);      // prefetch next stage into other buffer
            WAIT_ASYNC(4);              // stage s landed; stage s+1 may be in flight
        } else {
            WAIT_ASYNC(0);
        }
        __syncthreads();
        compute(cur);
        __syncthreads();                // all waves done reading buf before reuse
    }
#else
    for (int s = 0; s < nstage; ++s) {
        const int kk = s * 32;
        {
            const float4* pa = reinterpret_cast<const float4*>(Arow + kk);
            float4 a0 = pa[0], a1 = pa[1];
            *reinterpret_cast<float4*>(&sA[0][ldr][ldc])     = a0;
            *reinterpret_cast<float4*>(&sA[0][ldr][ldc + 8]) = a1;
        }
        {
            const float4* pb = reinterpret_cast<const float4*>(Brow + kk);
            float4 b0 = pb[0], b1 = pb[1];
            *reinterpret_cast<float4*>(&sB[0][ldr][ldc])     = b0;
            *reinterpret_cast<float4*>(&sB[0][ldr][ldc + 8]) = b1;
        }
        __syncthreads();
        compute(0);
        __syncthreads();
    }
#endif

    // ---- epilogue: D layout — lane L: VGPR r -> M = r + (L>=16?8:0), N = L%16 ----
    #pragma unroll
    for (int i = 0; i < 2; ++i) {
        #pragma unroll
        for (int j = 0; j < 2; ++j) {
            #pragma unroll
            for (int r = 0; r < 8; ++r) {
                int m = tileM + wm + i * 16 + lh * 8 + r;
                int n = tileN + wn + j * 16 + l16;
                if (m < M && n < N) {
                    float v = acc[i][j][r];
                    C[(size_t)m * N + n] = v;
                    if (Ch) Ch[(size_t)m * N + n] = (h16)v;
                }
            }
        }
    }
}

// =====================================================================
// Elementwise / sequential kernels (fp32 math)
// =====================================================================
__global__ void embed_kernel(const int* __restrict__ ids, const float* __restrict__ emb,
                             float* __restrict__ x32, h16* __restrict__ xh)
{
    int idx = blockIdx.x * blockDim.x + threadIdx.x;
    if (idx >= NTOK * DIM) return;
    int t = idx / DIM, d = idx - t * DIM;
    float v = emb[(size_t)ids[t] * DIM + d];
    x32[idx] = v;
    xh[idx]  = (h16)v;
}

// fp32 weight [N,K] -> f16 [N,Kp] with zero K-padding
__global__ void convert_w_kernel(const float* __restrict__ src, h16* __restrict__ dst,
                                 int N, int K, int Kp)
{
    int idx = blockIdx.x * blockDim.x + threadIdx.x;
    if (idx >= N * Kp) return;
    int n = idx / Kp, k = idx - n * Kp;
    dst[idx] = (h16)((k < K) ? src[(size_t)n * K + k] : 0.0f);
}

// depthwise causal conv (width 4) + SiLU, reading x_in half of xz
__global__ void conv_silu_kernel(const float* __restrict__ xz,
                                 const float* __restrict__ cw, const float* __restrict__ cb,
                                 float* __restrict__ xs32, h16* __restrict__ xsh)
{
    int idx = blockIdx.x * blockDim.x + threadIdx.x;
    if (idx >= NTOK * D_INNER) return;
    int m = idx / D_INNER, c = idx - m * D_INNER;
    int l = m & (SEQ - 1);
    float a = cb[c];
    #pragma unroll
    for (int k = 0; k < D_CONV; ++k) {
        int lk = l + k - (D_CONV - 1);
        if (lk >= 0)
            a += xz[(size_t)(m + k - (D_CONV - 1)) * XDIM + c] * cw[c * D_CONV + k];
    }
    float s = silu_f(a);
    xs32[idx] = s;
    xsh[idx]  = (h16)s;
}

// slice dbl[:, :48] into f16 padded [M,64] for the dt GEMM
__global__ void dtin_kernel(const float* __restrict__ dbl, h16* __restrict__ dtin)
{
    int idx = blockIdx.x * blockDim.x + threadIdx.x;
    if (idx >= NTOK * DT_RANK_P) return;
    int m = idx / DT_RANK_P, j = idx - m * DT_RANK_P;
    dtin[idx] = (h16)((j < DT_RANK) ? dbl[(size_t)m * DBLDIM + j] : 0.0f);
}

// dt = softplus(dtp + b_dt), in place
__global__ void softplus_kernel(float* __restrict__ dtp, const float* __restrict__ b_dt)
{
    int idx = blockIdx.x * blockDim.x + threadIdx.x;
    if (idx >= NTOK * D_INNER) return;
    int c = idx % D_INNER;
    float v = dtp[idx] + b_dt[c];
    dtp[idx] = (v > 20.0f) ? v : log1pf(__expf(v));
}

// selective scan: one thread per (batch, channel); fused +x*D and *silu(z)
__global__ void __launch_bounds__(256)
scan_kernel(const float* __restrict__ dt,   const float* __restrict__ xs,
            const float* __restrict__ dbl,  const float* __restrict__ xz,
            const float* __restrict__ A_log, const float* __restrict__ Dp,
            h16* __restrict__ yh)
{
    int c = blockIdx.x * 256 + threadIdx.x;
    int b = blockIdx.y;
    if (c >= D_INNER) return;

    float A[D_STATE], h[D_STATE];
    #pragma unroll
    for (int s = 0; s < D_STATE; ++s) {
        A[s] = -__expf(A_log[(size_t)c * D_STATE + s]);
        h[s] = 0.0f;
    }
    const float Dc = Dp[c];

    for (int l = 0; l < SEQ; ++l) {
        size_t m = (size_t)b * SEQ + l;
        float dtv = dt[m * D_INNER + c];
        float xv  = xs[m * D_INNER + c];
        const float* Bt = dbl + m * DBLDIM + DT_RANK;
        const float* Ct = Bt + D_STATE;
        float dx = dtv * xv;
        float yv = 0.0f;
        #pragma unroll
        for (int s = 0; s < D_STATE; ++s) {
            h[s] = h[s] * __expf(dtv * A[s]) + dx * Bt[s];
            yv  += h[s] * Ct[s];
        }
        yv += xv * Dc;
        float zv  = xz[m * XDIM + D_INNER + c];
        float out = yv * silu_f(zv);
        yh[m * D_INNER + c] = (h16)out;
    }
}

// LayerNorm over DIM, writes f16 staging for the head GEMM
__global__ void __launch_bounds__(256)
layernorm_kernel(const float* __restrict__ x, const float* __restrict__ g,
                 const float* __restrict__ bta, h16* __restrict__ outh)
{
    __shared__ float red[256];
    int m = blockIdx.x;
    const float* row = x + (size_t)m * DIM;

    float s = 0.0f;
    for (int d = threadIdx.x; d < DIM; d += 256) s += row[d];
    red[threadIdx.x] = s; __syncthreads();
    for (int off = 128; off > 0; off >>= 1) {
        if (threadIdx.x < off) red[threadIdx.x] += red[threadIdx.x + off];
        __syncthreads();
    }
    float mu = red[0] / (float)DIM;
    __syncthreads();

    float v = 0.0f;
    for (int d = threadIdx.x; d < DIM; d += 256) { float t = row[d] - mu; v += t * t; }
    red[threadIdx.x] = v; __syncthreads();
    for (int off = 128; off > 0; off >>= 1) {
        if (threadIdx.x < off) red[threadIdx.x] += red[threadIdx.x + off];
        __syncthreads();
    }
    float rstd = rsqrtf(red[0] / (float)DIM + LN_EPS);

    for (int d = threadIdx.x; d < DIM; d += 256)
        outh[(size_t)m * DIM + d] = (h16)((row[d] - mu) * rstd * g[d] + bta[d]);
}

// =====================================================================
// host-side orchestration
// =====================================================================
extern "C" void kernel_launch(void* const* d_in, const int* in_sizes, int n_in,
                              void* d_out, int out_size, void* d_ws, size_t ws_size,
                              hipStream_t stream)
{
    (void)in_sizes; (void)n_in; (void)out_size; (void)ws_size;

    const int*   ids    = (const int*)  d_in[0];
    const float* emb    = (const float*)d_in[1];
    const float* W_in   = (const float*)d_in[2];
    const float* conv_w = (const float*)d_in[3];
    const float* conv_b = (const float*)d_in[4];
    const float* W_x    = (const float*)d_in[5];
    const float* W_dt   = (const float*)d_in[6];
    const float* b_dt   = (const float*)d_in[7];
    const float* A_log  = (const float*)d_in[8];
    const float* Dp     = (const float*)d_in[9];
    const float* W_out  = (const float*)d_in[10];
    const float* ln_g   = (const float*)d_in[11];
    const float* ln_b   = (const float*)d_in[12];
    const float* W_head = (const float*)d_in[13];
    float* out = (float*)d_out;

    // ---- workspace carve-out (256B aligned) ----
    char* ws_cur = (char*)d_ws;
    auto walloc = [&](size_t bytes) -> char* {
        char* p = ws_cur;
        ws_cur += (bytes + 255) & ~(size_t)255;
        return p;
    };
    float* x32   = (float*)walloc(sizeof(float) * (size_t)NTOK * DIM);
    h16*   xh    = (h16*)  walloc(sizeof(h16)   * (size_t)NTOK * DIM);
    float* xz32  = (float*)walloc(sizeof(float) * (size_t)NTOK * XDIM);
    float* xs32  = (float*)walloc(sizeof(float) * (size_t)NTOK * D_INNER);
    h16*   xsh   = (h16*)  walloc(sizeof(h16)   * (size_t)NTOK * D_INNER);
    float* dbl32 = (float*)walloc(sizeof(float) * (size_t)NTOK * DBLDIM);
    h16*   dtin  = (h16*)  walloc(sizeof(h16)   * (size_t)NTOK * DT_RANK_P);
    float* dt32  = (float*)walloc(sizeof(float) * (size_t)NTOK * D_INNER);
    h16*   yh    = (h16*)  walloc(sizeof(h16)   * (size_t)NTOK * D_INNER);
    h16*   Winh  = (h16*)  walloc(sizeof(h16)   * (size_t)XDIM * DIM);
    h16*   Wxh   = (h16*)  walloc(sizeof(h16)   * (size_t)DBLDIM * D_INNER);
    h16*   Wdth  = (h16*)  walloc(sizeof(h16)   * (size_t)D_INNER * DT_RANK_P);
    h16*   Wouth = (h16*)  walloc(sizeof(h16)   * (size_t)DIM * D_INNER);
    h16*   Whdh  = (h16*)  walloc(sizeof(h16)   * (size_t)VOCAB * DIM);

    auto blocks = [](size_t n) { return (unsigned)((n + 255) / 256); };
    auto gemm = [&](const h16* A, const h16* B, float* C, h16* Ch, int M, int N, int Kp) {
        dim3 grid((N + 63) / 64, (M + 63) / 64);
        wmma_gemm_kernel<<<grid, 128, 0, stream>>>(A, B, C, Ch, M, N, Kp);
    };

    // ---- embedding ----
    embed_kernel<<<blocks((size_t)NTOK * DIM), 256, 0, stream>>>(ids, emb, x32, xh);

    // ---- mamba blocks ----
    for (int layer = 0; layer < DEPTH; ++layer) {
        const float* Wi = W_in   + (size_t)layer * XDIM * DIM;
        const float* cw = conv_w + (size_t)layer * D_INNER * D_CONV;
        const float* cb = conv_b + (size_t)layer * D_INNER;
        const float* Wx = W_x    + (size_t)layer * DBLDIM * D_INNER;
        const float* Wd = W_dt   + (size_t)layer * D_INNER * DT_RANK;
        const float* bd = b_dt   + (size_t)layer * D_INNER;
        const float* Al = A_log  + (size_t)layer * D_INNER * D_STATE;
        const float* Dl = Dp     + (size_t)layer * D_INNER;
        const float* Wo = W_out  + (size_t)layer * DIM * D_INNER;

        // in-projection: xz[M,3072] = x[M,768] @ W_in^T
        convert_w_kernel<<<blocks((size_t)XDIM * DIM), 256, 0, stream>>>(Wi, Winh, XDIM, DIM, DIM);
        gemm(xh, Winh, xz32, nullptr, NTOK, XDIM, DIM);

        // conv + SiLU -> x_ssm
        conv_silu_kernel<<<blocks((size_t)NTOK * D_INNER), 256, 0, stream>>>(xz32, cw, cb, xs32, xsh);

        // x-projection: dbl[M,80] = x_ssm @ W_x^T
        convert_w_kernel<<<blocks((size_t)DBLDIM * D_INNER), 256, 0, stream>>>(Wx, Wxh, DBLDIM, D_INNER, D_INNER);
        gemm(xsh, Wxh, dbl32, nullptr, NTOK, DBLDIM, D_INNER);

        // dt projection: dt[M,1536] = dbl[:, :48] @ W_dt^T  (K padded 48->64)
        dtin_kernel<<<blocks((size_t)NTOK * DT_RANK_P), 256, 0, stream>>>(dbl32, dtin);
        convert_w_kernel<<<blocks((size_t)D_INNER * DT_RANK_P), 256, 0, stream>>>(Wd, Wdth, D_INNER, DT_RANK, DT_RANK_P);
        gemm(dtin, Wdth, dt32, nullptr, NTOK, D_INNER, DT_RANK_P);
        softplus_kernel<<<blocks((size_t)NTOK * D_INNER), 256, 0, stream>>>(dt32, bd);

        // selective scan (fused +x*D and *silu(z)) -> yh (f16)
        {
            dim3 grid(D_INNER / 256, BATCH);
            scan_kernel<<<grid, 256, 0, stream>>>(dt32, xs32, dbl32, xz32, Al, Dl, yh);
        }

        // out-projection: x_next[M,768] = y @ W_out^T ; also emit f16 for next layer
        convert_w_kernel<<<blocks((size_t)DIM * D_INNER), 256, 0, stream>>>(Wo, Wouth, DIM, D_INNER, D_INNER);
        gemm(yh, Wouth, x32, xh, NTOK, DIM, D_INNER);
    }

    // ---- final LayerNorm + LM head ----
    layernorm_kernel<<<NTOK, 256, 0, stream>>>(x32, ln_g, ln_b, xh);
    convert_w_kernel<<<blocks((size_t)VOCAB * DIM), 256, 0, stream>>>(W_head, Whdh, VOCAB, DIM, DIM);
    gemm(xh, Whdh, out, nullptr, NTOK, VOCAB, DIM);
}
